// FullAttention_3375844295096
// MI455X (gfx1250) — compile-verified, matrix operand order checked by reading, not run on hardware
//
#include <hip/hip_runtime.h>

typedef __attribute__((ext_vector_type(16))) _Float16 v16h;
typedef __attribute__((ext_vector_type(8)))  float    v8f;
typedef __attribute__((ext_vector_type(4)))  float    f32x4;

#define S_LEN 1024
#define DH    64
#define VROW  68   // 64 data dwords + 4 pad dwords per LDS row (TDM pad feature)

__device__ __forceinline__ int iabs(int x) { return x < 0 ? -x : x; }

#if defined(__has_builtin)
#if __has_builtin(__builtin_amdgcn_tensor_load_to_lds) && __has_builtin(__builtin_amdgcn_s_wait_tensorcnt)
#define USE_TDM 1
#endif
#endif

#ifdef USE_TDM
// Issue one TDM 2D tile load: 32 rows x 64 cols of f32, row stride 64 floats,
// LDS rows padded to 68 floats via pad_interval=64dw / pad_amount=4dw.
__device__ __forceinline__ void tdm_load_tile(const void* gaddr, unsigned lds_off) {
  typedef unsigned v4u __attribute__((ext_vector_type(4)));
  typedef int      v8i __attribute__((ext_vector_type(8)));
  typedef int      v4i __attribute__((ext_vector_type(4)));
  unsigned long long ga = (unsigned long long)(size_t)gaddr;
  // D# group0: count=1 | lds_addr | global_addr[56:0] | type=2
  v4u g0 = { 1u, lds_off, (unsigned)ga,
             ((unsigned)(ga >> 32) & 0x01ffffffu) | 0x80000000u };
  // D# group1: data_size=4B(code2), pad_enable, pad_interval=64dw(code5),
  // pad_amount=4dw(code3); tensor 64x32, tile 64x32, dim0 stride 64.
  v8i g1 = { (int)((2u << 16) | (1u << 20) | (5u << 22) | (3u << 25)),
             (int)(64u << 16),   // tensor_dim0 = 64   (bits 79:48)
             (int)(32u << 16),   // tensor_dim1 = 32   (bits 111:80)
             (int)(64u << 16),   // tile_dim0   = 64   (bits 127:112)
             (int)32,            // tile_dim1   = 32
             (int)64,            // tensor_dim0_stride = 64
             0, 0 };
  v4i z4 = { 0, 0, 0, 0 };
  v8i z8 = { 0, 0, 0, 0, 0, 0, 0, 0 };
  // 6-arg toolchain variant: (g0, g1, g2, g3, g4, cpol)
  __builtin_amdgcn_tensor_load_to_lds(g0, g1, z4, z4, z8, 0);
}
#endif

__global__ __launch_bounds__(256)
void fa3d_wmma_kernel(const float* __restrict__ Q, const float* __restrict__ K,
                      const float* __restrict__ V, float* __restrict__ O) {
  __shared__ __attribute__((aligned(16))) float ldsK[2][32 * VROW];
  __shared__ __attribute__((aligned(16))) float ldsV[2][32 * VROW];

  const int lane = threadIdx.x & 31;
  const int wave = threadIdx.x >> 5;
  const int qb   = blockIdx.x & 7;    // 8 query blocks of 128 rows
  const int bh   = blockIdx.x >> 3;   // 64 (b,h) pairs
  const int q0   = qb * 128 + wave * 16;
  const int ln   = lane & 15;
  const int hi   = (lane >> 4) << 3;  // 0 for lanes 0-15, 8 for lanes 16-31

  const size_t base = (size_t)bh * S_LEN * DH;
  const int qy = q0 + ln;             // this lane's query row (fixed)
  const int qt = qy >> 8, qh = (qy >> 4) & 15, qw = qy & 15;

  // ---- Q as B-operand fragments (k-dim = d), scaled by 1/sqrt(D).
  // f16 operand layout: element i <-> k-dim index (i<8 ? i : i+8) + hi
  v16h bq0, bq1;
  {
    const float* qp = Q + base + (size_t)qy * DH;
    const float sc = 0.125f;  // 1/sqrt(64)
#pragma unroll
    for (int i = 0; i < 8; ++i) {
      bq0[i]     = (_Float16)(qp[hi + i]      * sc);
      bq0[i + 8] = (_Float16)(qp[16 + hi + i] * sc);
      bq1[i]     = (_Float16)(qp[32 + hi + i] * sc);
      bq1[i + 8] = (_Float16)(qp[48 + hi + i] * sc);
    }
  }

  v8f oacc[4];
#pragma unroll
  for (int dt = 0; dt < 4; ++dt) oacc[dt] = (v8f){};
  float m_run = -__builtin_inff();
  float l_run = 0.f;

  const int nblk_w = (q0 + 47) >> 5;           // this wave's causal limit
  const int nblk_u = (qb * 128 + 159) >> 5;    // block-uniform trip count

#ifdef USE_TDM
  if (wave == 0) {                             // prologue: stage block 0
    tdm_load_tile(K + base, (unsigned)(size_t)&ldsK[0][0]);
    tdm_load_tile(V + base, (unsigned)(size_t)&ldsV[0][0]);
  }
#endif

  for (int blk = 0; blk < nblk_u; ++blk) {
    const int kb  = blk << 5;
    const int buf = blk & 1;

#ifdef USE_TDM
    if (wave == 0) {
      if (blk + 1 < nblk_u) {                  // stage next block into buf^1
        const size_t goff = base + (size_t)(kb + 32) * DH;
        tdm_load_tile(K + goff, (unsigned)(size_t)&ldsK[buf ^ 1][0]);
        tdm_load_tile(V + goff, (unsigned)(size_t)&ldsV[buf ^ 1][0]);
        __builtin_amdgcn_s_wait_tensorcnt((short)2);  // buf ready, next in flight
      } else {
        __builtin_amdgcn_s_wait_tensorcnt((short)0);
      }
    }
    __syncthreads();                           // buf visible to all waves
#else
    __syncthreads();                           // previous reads done
    for (int e = threadIdx.x; e < 2048; e += 256) {
      const int r = e >> 6, c = e & 63;
      ldsK[buf][r * VROW + c] = K[base + (size_t)(kb + r) * DH + c];
      ldsV[buf][r * VROW + c] = V[base + (size_t)(kb + r) * DH + c];
    }
    __syncthreads();
#endif

    if (blk < nblk_w) {
      // ---- GEMM1: S^T(16k x 16q) = K_tile * Q^T, two 16-key subtiles ------
      v8f s[2];
#pragma unroll
      for (int kt = 0; kt < 2; ++kt) {
        const f32x4* kr = (const f32x4*)&ldsK[buf][(kt * 16 + ln) * VROW];
        const int c0 = hi >> 2;                // {0,2}
        f32x4 ra = kr[c0],     rb = kr[c0 + 1];   // cols hi   .. hi+7
        f32x4 rc = kr[c0 + 4], rd = kr[c0 + 5];   // cols 16+hi.. +7
        f32x4 re = kr[c0 + 8], rf = kr[c0 + 9];   // cols 32+hi.. +7
        f32x4 rg = kr[c0 +12], rh = kr[c0 +13];   // cols 48+hi.. +7
        v16h a0, a1;
#pragma unroll
        for (int i = 0; i < 4; ++i) {
          a0[i]      = (_Float16)ra[i];  a0[i + 4]  = (_Float16)rb[i];
          a0[i + 8]  = (_Float16)rc[i];  a0[i + 12] = (_Float16)rd[i];
          a1[i]      = (_Float16)re[i];  a1[i + 4]  = (_Float16)rf[i];
          a1[i + 8]  = (_Float16)rg[i];  a1[i + 12] = (_Float16)rh[i];
        }
        v8f c = (v8f){};
        c = __builtin_amdgcn_wmma_f32_16x16x32_f16(false, a0, false, bq0,
                                                   (short)0, c, false, false);
        c = __builtin_amdgcn_wmma_f32_16x16x32_f16(false, a1, false, bq1,
                                                   (short)0, c, false, false);
        s[kt] = c;
      }

      // ---- causal + distance mask, online softmax (per-lane query col) ----
      float p[16];
      float mb = -__builtin_inff();
#pragma unroll
      for (int kt = 0; kt < 2; ++kt) {
#pragma unroll
        for (int r = 0; r < 8; ++r) {
          const int ky = kb + kt * 16 + r + hi;  // D layout: M = r + hi
          float sv;
          if (ky <= qy) {
            const int kt_ = ky >> 8, kh = (ky >> 4) & 15, kw = ky & 15;
            const int dist = iabs(qt - kt_) + iabs(qh - kh) + iabs(qw - kw);
            sv = s[kt][r] * __expf((float)dist * (-1.0f / 33.0f));
          } else {
            sv = -__builtin_inff();
          }
          p[kt * 8 + r] = sv;
          mb = fmaxf(mb, sv);
        }
      }
      mb = fmaxf(mb, __shfl_xor(mb, 16, 32));  // partner half: other 8 keys
      const float m_new = fmaxf(m_run, mb);
      float lp = 0.f;
#pragma unroll
      for (int i = 0; i < 16; ++i) { p[i] = __expf(p[i] - m_new); lp += p[i]; }
      lp += __shfl_xor(lp, 16, 32);
      const float alpha = __expf(m_run - m_new);
      l_run = l_run * alpha + lp;
      m_run = m_new;
#pragma unroll
      for (int dt = 0; dt < 4; ++dt)
#pragma unroll
        for (int r = 0; r < 8; ++r) oacc[dt][r] *= alpha;

      // ---- pack P into B-operand: element i -> k = (i<8?i:i+8)+hi ---------
      v16h pb;
#pragma unroll
      for (int i = 0; i < 8; ++i) {
        pb[i]     = (_Float16)p[i];
        pb[i + 8] = (_Float16)p[8 + i];
      }

      // ---- GEMM2: O^T(16d x 16q) += V^T(16d x 32k) * P^T(32k x 16q) -------
#pragma unroll
      for (int dt = 0; dt < 4; ++dt) {
        const float* vb = &ldsV[buf][dt * 16 + ln];
        v16h av;
#pragma unroll
        for (int i = 0; i < 8; ++i) {
          av[i]     = (_Float16)vb[(i + hi) * VROW];
          av[i + 8] = (_Float16)vb[(16 + i + hi) * VROW];
        }
        oacc[dt] = __builtin_amdgcn_wmma_f32_16x16x32_f16(false, av, false, pb,
                                                          (short)0, oacc[dt],
                                                          false, false);
      }
    }
    __syncthreads();                 // all reads of buf done before overwrite
  }

  // ---- normalize and store: d = dt*16 + r + hi, row = qy -------------------
  const float inv_l = 1.0f / l_run;
  float* op = O + base + (size_t)qy * DH;
#pragma unroll
  for (int dt = 0; dt < 4; ++dt)
#pragma unroll
    for (int r = 0; r < 8; ++r)
      op[dt * 16 + r + hi] = oacc[dt][r] * inv_l;
}

extern "C" void kernel_launch(void* const* d_in, const int* in_sizes, int n_in,
                              void* d_out, int out_size, void* d_ws, size_t ws_size,
                              hipStream_t stream) {
  const float* Q = (const float*)d_in[0];
  const float* K = (const float*)d_in[1];
  const float* V = (const float*)d_in[2];
  float* O = (float*)d_out;
  dim3 grid(64 * 8), block(256);
  fa3d_wmma_kernel<<<grid, block, 0, stream>>>(Q, K, V, O);
}
